// QLinear_64347200028815
// MI455X (gfx1250) — matrix operand order repaired and checked by reading, compiled, not run on hardware
//
#include <hip/hip_runtime.h>

// ---- CDNA5 WMMA vector types (probe-verified signatures) ----
typedef __attribute__((ext_vector_type(16))) __bf16 v16bf;
typedef __attribute__((ext_vector_type(8)))  __bf16 v8bf;
typedef __attribute__((ext_vector_type(8)))  float  v8f;
typedef __attribute__((ext_vector_type(4)))  float  v4f;
// gcc-style vector to match the async-LDS builtin's parameter type exactly
typedef int v4i_g __attribute__((vector_size(16)));

#define IN_CH 1024
#define K_DIM 4096
#define N_DIM 4096
#define BM 128
#define BN 128
#define BK 64
#define LDST 88   // LDS row stride in bf16: 176B -> 16B-aligned fragments, low bank conflicts
#define NT (K_DIM / BK)   // 64 k-tiles

#define GLOBAL_AS __attribute__((address_space(1)))
#define LDS_AS    __attribute__((address_space(3)))

#if __has_builtin(__builtin_amdgcn_global_load_async_to_lds_b128)
#define USE_ASYNC_LDS 1
#else
#define USE_ASYNC_LDS 0
#endif

__device__ __forceinline__ void async_copy16(const void* g, void* l) {
#if USE_ASYNC_LDS
    __builtin_amdgcn_global_load_async_to_lds_b128(
        (GLOBAL_AS v4i_g*)g, (LDS_AS v4i_g*)l, 0, 0);
#else
    (void)g; (void)l;
#endif
}

template<int N>
__device__ __forceinline__ void wait_asynccnt() {
#if __has_builtin(__builtin_amdgcn_s_wait_asynccnt)
    __builtin_amdgcn_s_wait_asynccnt(N);
#else
    asm volatile("s_wait_asynccnt %0" :: "i"(N) : "memory");
#endif
}

// Hamilton tables: B[kk][n] = sign[qk][qn] * comp[cidx[qk][qn]][kk%1024][n%1024]
__constant__ int   c_cidx[4][4] = { {0,1,2,3}, {1,0,3,2}, {2,3,0,1}, {3,2,1,0} };
__constant__ float c_csgn[4][4] = { { 1.f, 1.f, 1.f, 1.f},
                                    {-1.f, 1.f, 1.f,-1.f},
                                    {-1.f,-1.f, 1.f, 1.f},
                                    {-1.f, 1.f,-1.f, 1.f} };
// slot = cidx*2 + (sign<0): index into pre-baked signed bf16 weights
__constant__ int   c_slot[4][4] = { {0,2,4,6}, {3,0,6,5}, {5,7,0,2}, {7,4,3,0} };

// ------------------------------------------------------------------
// Prep kernel 1: x fp32 -> bf16 (same layout), 8 elems/thread
// ------------------------------------------------------------------
__global__ __launch_bounds__(256) void conv_x_bf16(const float* __restrict__ x,
                                                   __bf16* __restrict__ xb)
{
    const size_t i = ((size_t)blockIdx.x * 256 + threadIdx.x) * 8;
    v4f f0 = *(const v4f*)(x + i);
    v4f f1 = *(const v4f*)(x + i + 4);
    v8bf h;
    h[0] = (__bf16)f0[0]; h[1] = (__bf16)f0[1]; h[2] = (__bf16)f0[2]; h[3] = (__bf16)f0[3];
    h[4] = (__bf16)f1[0]; h[5] = (__bf16)f1[1]; h[6] = (__bf16)f1[2]; h[7] = (__bf16)f1[3];
    *(v8bf*)(xb + i) = h;
}

// ------------------------------------------------------------------
// Prep kernel 2: build 8 signed, transposed bf16 weights:
//   Ws[slot][n][k] = (slot&1 ? -1 : +1) * comp[slot>>1][k][n]
// ------------------------------------------------------------------
__global__ __launch_bounds__(256) void prep_w_bf16(
    const float* __restrict__ rw, const float* __restrict__ iw,
    const float* __restrict__ jw, const float* __restrict__ kw,
    __bf16* __restrict__ dst)
{
    const int slot = blockIdx.y;
    const int ci   = slot >> 1;
    const float sg = (slot & 1) ? -1.f : 1.f;
    const float* src = (ci == 0) ? rw : (ci == 1) ? iw : (ci == 2) ? jw : kw;

    const int tn = (blockIdx.x & 15) * 64;
    const int tk = (blockIdx.x >> 4) * 64;

    __shared__ float tile[64][65];

    const int r  = threadIdx.x >> 2;
    const int c4 = threadIdx.x & 3;

    const float* sp = src + (size_t)(tk + r) * IN_CH + tn + c4 * 16;
#pragma unroll
    for (int i = 0; i < 4; ++i) {
        v4f v = *(const v4f*)(sp + i * 4);
        tile[r][c4 * 16 + i * 4 + 0] = v[0];
        tile[r][c4 * 16 + i * 4 + 1] = v[1];
        tile[r][c4 * 16 + i * 4 + 2] = v[2];
        tile[r][c4 * 16 + i * 4 + 3] = v[3];
    }
    __syncthreads();

    __bf16* dp = dst + (size_t)slot * (IN_CH * IN_CH)
                     + (size_t)(tn + r) * IN_CH + tk + c4 * 16;
    v8bf h0, h1;
#pragma unroll
    for (int j = 0; j < 8; ++j) h0[j] = (__bf16)(sg * tile[c4 * 16 + j][r]);
#pragma unroll
    for (int j = 0; j < 8; ++j) h1[j] = (__bf16)(sg * tile[c4 * 16 + 8 + j][r]);
    *(v8bf*)dp       = h0;
    *(v8bf*)(dp + 8) = h1;
}

// ------------------------------------------------------------------
// Main GEMM: out = x @ W_hamilton + bias  (M=16384, K=N=4096)
// Double-buffered LDS; async global->LDS copies on the all-bf16 path.
// ------------------------------------------------------------------
template<bool ABF, bool BBF>
__global__ __launch_bounds__(256) void qlinear_wmma_bf16(
    const float*  __restrict__ x,
    const __bf16* __restrict__ xb,
    const float*  __restrict__ rw,
    const float*  __restrict__ iw,
    const float*  __restrict__ jw,
    const float*  __restrict__ kw,
    const __bf16* __restrict__ wsW,
    const float*  __restrict__ bias,
    float*        __restrict__ out)
{
    constexpr bool ASYNCF = (ABF && BBF) && (USE_ASYNC_LDS != 0);

    __shared__ __bf16 As[2][BM * LDST];   // [m][k]
    __shared__ __bf16 Bt[2][BN * LDST];   // [n][k]

    const int tid   = threadIdx.x;
    const int lane  = tid & 31;
    const int l15   = lane & 15;
    const int lhalf = lane >> 4;
    const int wid   = tid >> 5;

    const int n0 = blockIdx.x * BN;
    const int m0 = blockIdx.y * BM;
    const int qn = n0 >> 10;
    const int nq = n0 & 1023;

    const int wave_m = (wid >> 2) * 64;
    const int wave_n = (wid & 3)  * 32;

    const int a_row  = tid >> 1;           // 0..127
    const int a_col0 = (tid & 1) * 32;     // 0 or 32
    const int b_n    = BBF ? (tid >> 1) : (tid & 127);
    const int b_k0   = BBF ? ((tid & 1) * 32) : ((tid >> 7) * 32);

    v8f acc[4][2];
#pragma unroll
    for (int mt = 0; mt < 4; ++mt)
#pragma unroll
        for (int nt = 0; nt < 2; ++nt)
            acc[mt][nt] = (v8f)0.f;

    __bf16* asL[2] = { &As[0][a_row * LDST + a_col0], &As[1][a_row * LDST + a_col0] };
    __bf16* bsL[2] = { &Bt[0][b_n  * LDST + b_k0],    &Bt[1][b_n  * LDST + b_k0]    };
    const float*  agf = ABF ? nullptr : (x  + (size_t)(m0 + a_row) * K_DIM + a_col0);
    const __bf16* agh = ABF ? (xb + (size_t)(m0 + a_row) * K_DIM + a_col0) : nullptr;

    // ---- tile fill: t = k-tile index (k0 = t*BK), buf = LDS buffer ----
    auto fill = [&](int t, int buf) {
        // A tile
        if constexpr (ABF) {
            const __bf16* ag = agh + (size_t)t * BK;
            if constexpr (ASYNCF) {
#pragma unroll
                for (int i = 0; i < 4; ++i)
                    async_copy16(ag + i * 8, asL[buf] + i * 8);
            } else {
#pragma unroll
                for (int i = 0; i < 4; ++i)
                    *(v8bf*)(asL[buf] + i * 8) = *(const v8bf*)(ag + i * 8);
            }
        } else {
            const float* ag = agf + (size_t)t * BK;
#pragma unroll
            for (int i = 0; i < 4; ++i) {
                v4f f0 = *(const v4f*)(ag + i * 8);
                v4f f1 = *(const v4f*)(ag + i * 8 + 4);
                v8bf h;
                h[0] = (__bf16)f0[0]; h[1] = (__bf16)f0[1];
                h[2] = (__bf16)f0[2]; h[3] = (__bf16)f0[3];
                h[4] = (__bf16)f1[0]; h[5] = (__bf16)f1[1];
                h[6] = (__bf16)f1[2]; h[7] = (__bf16)f1[3];
                *(v8bf*)(asL[buf] + i * 8) = h;
            }
        }
        // B tile (transposed [n][k])
        const int qk = t >> 4;
        const int kk = (t & 15) * BK;
        if constexpr (BBF) {
            const int slot = c_slot[qk][qn];
            const __bf16* wg = wsW + (size_t)slot * (IN_CH * IN_CH)
                                   + (size_t)(nq + b_n) * IN_CH + b_k0 + kk;
            if constexpr (ASYNCF) {
#pragma unroll
                for (int i = 0; i < 4; ++i)
                    async_copy16(wg + i * 8, bsL[buf] + i * 8);
            } else {
#pragma unroll
                for (int i = 0; i < 4; ++i)
                    *(v8bf*)(bsL[buf] + i * 8) = *(const v8bf*)(wg + i * 8);
            }
        } else {
            const int   ci  = c_cidx[qk][qn];
            const float sgn = c_csgn[qk][qn];
            const float* wg = (((ci == 0) ? rw : (ci == 1) ? iw : (ci == 2) ? jw : kw))
                              + (size_t)(kk + b_k0) * IN_CH + nq + b_n;
#pragma unroll
            for (int i = 0; i < 4; ++i) {
                v8bf h;
#pragma unroll
                for (int j = 0; j < 8; ++j)
                    h[j] = (__bf16)(sgn * wg[(size_t)(i * 8 + j) * IN_CH]);
                *(v8bf*)(bsL[buf] + i * 8) = h;
            }
        }
    };

    fill(0, 0);

#pragma unroll 1
    for (int t = 0; t < NT; ++t) {
        const int cur = t & 1;

        if (t + 1 < NT) {
            fill(t + 1, cur ^ 1);        // buffer last read two barriers ago
            if constexpr (ASYNCF) wait_asynccnt<8>();   // tile t's 8 asyncs done (in-order)
        } else {
            if constexpr (ASYNCF) wait_asynccnt<0>();
        }
        __syncthreads();

        const __bf16* Ab = As[cur];
        const __bf16* Bb = Bt[cur];
#pragma unroll
        for (int ks = 0; ks < BK; ks += 32) {
            v16bf af[4];
#pragma unroll
            for (int mt = 0; mt < 4; ++mt) {
                const __bf16* ap = &Ab[(wave_m + mt * 16 + l15) * LDST + ks + lhalf * 8];
                v8bf lo = *(const v8bf*)ap;
                v8bf hi = *(const v8bf*)(ap + 16);
                af[mt] = __builtin_shufflevector(lo, hi,
                    0,1,2,3,4,5,6,7,8,9,10,11,12,13,14,15);
            }
            v16bf bfv[2];
#pragma unroll
            for (int nt = 0; nt < 2; ++nt) {
                const __bf16* bp = &Bb[(wave_n + nt * 16 + l15) * LDST + ks + lhalf * 8];
                v8bf lo = *(const v8bf*)bp;
                v8bf hi = *(const v8bf*)(bp + 16);
                bfv[nt] = __builtin_shufflevector(lo, hi,
                    0,1,2,3,4,5,6,7,8,9,10,11,12,13,14,15);
            }
#pragma unroll
            for (int mt = 0; mt < 4; ++mt)
#pragma unroll
                for (int nt = 0; nt < 2; ++nt)
                    acc[mt][nt] = __builtin_amdgcn_wmma_f32_16x16x32_bf16(
                        false, af[mt], false, bfv[nt],
                        (short)0, acc[mt][nt], false, false);
        }
        __syncthreads();
    }

    // ---------- epilogue ----------
#pragma unroll
    for (int nt = 0; nt < 2; ++nt) {
        const int n  = n0 + wave_n + nt * 16 + l15;
        const float bv = bias[n];
#pragma unroll
        for (int mt = 0; mt < 4; ++mt) {
            const int mbase = m0 + wave_m + mt * 16 + lhalf * 8;
#pragma unroll
            for (int i = 0; i < 8; ++i)
                out[(size_t)(mbase + i) * N_DIM + n] = acc[mt][nt][i] + bv;
        }
    }
}

extern "C" void kernel_launch(void* const* d_in, const int* in_sizes, int n_in,
                              void* d_out, int out_size, void* d_ws, size_t ws_size,
                              hipStream_t stream) {
    (void)n_in; (void)out_size;
    const float* x    = (const float*)d_in[0];
    const float* rw   = (const float*)d_in[1];
    const float* iw   = (const float*)d_in[2];
    const float* jw   = (const float*)d_in[3];
    const float* kwp  = (const float*)d_in[4];
    const float* bias = (const float*)d_in[5];
    float* out = (float*)d_out;

    const int M = in_sizes[0] / K_DIM;                       // 16384
    const size_t wbytes = (size_t)8 * IN_CH * IN_CH * 2;     // 16 MB signed bf16 weights
    const size_t xbytes = (size_t)in_sizes[0] * 2;           // 128 MB bf16 x

    const bool haveW = ws_size >= wbytes;
    const bool haveX = ws_size >= wbytes + xbytes;

    __bf16* wsW = (__bf16*)d_ws;
    __bf16* wsX = (__bf16*)((char*)d_ws + wbytes);

    if (haveW) {
        dim3 g(256, 8);
        prep_w_bf16<<<g, 256, 0, stream>>>(rw, iw, jw, kwp, wsW);
    }
    if (haveX) {
        conv_x_bf16<<<in_sizes[0] / 2048, 256, 0, stream>>>(x, wsX);
    }

    dim3 grid(N_DIM / BN, M / BM);                            // (32, 128)
    if (haveX && haveW)
        qlinear_wmma_bf16<true,  true ><<<grid, 256, 0, stream>>>(x, wsX, rw, iw, jw, kwp, wsW, bias, out);
    else if (haveW)
        qlinear_wmma_bf16<false, true ><<<grid, 256, 0, stream>>>(x, wsX, rw, iw, jw, kwp, wsW, bias, out);
    else
        qlinear_wmma_bf16<false, false><<<grid, 256, 0, stream>>>(x, wsX, rw, iw, jw, kwp, wsW, bias, out);
}
